// VectorQuantizerEMA_60138132078731
// MI455X (gfx1250) — matrix operand order repaired and checked by reading, not compiled, not run on hardware
//
#include <hip/hip_runtime.h>
#include <hip/hip_bf16.h>

// ---------------------------------------------------------------------------
// VQ-VAE nearest-codebook search on MI455X (gfx1250, wave32, WMMA).
// dist(n,k) = ||z_n||^2 + ||e_k||^2 - 2 z_n.e_k ; argmin_k only needs
// e2[k] - 2*dot, so ||z||^2 is dropped. Dot products via
// v_wmma_f32_16x16x32_bf16 with a bf16 hi/lo error-split (3 passes).
// ---------------------------------------------------------------------------

typedef __attribute__((ext_vector_type(16))) __bf16 v16bf;
typedef __attribute__((ext_vector_type(8)))  float  v8f;

#define N_PTS   131072
#define K_EMB   1024
#define D_DIM   64
#define BETA_C  10.0f
#define N_CTILE (K_EMB / 16)              // 64 column tiles of 16 codes
#define MBLK    2                         // 16-row tiles per wave (B reuse)
#define WAVES   8                         // waves per block (256 threads)
#define ROWS_PER_BLOCK (MBLK * 16 * WAVES)        // 256 rows
#define N_BLOCKS       (N_PTS / ROWS_PER_BLOCK)   // 512
#define N_WAVES_TOTAL  (N_BLOCKS * WAVES)         // 4096

// workspace layout (byte offsets)
#define WS_BHI   0
#define WS_BLO   (WS_BHI + K_EMB * D_DIM * 2)
#define WS_E2    (WS_BLO + K_EMB * D_DIM * 2)
#define WS_CNT   (WS_E2  + K_EMB * 4)
#define WS_PART  (WS_CNT + K_EMB * 4)

// ---------------------------------------------------------------------------
// Prep: swizzle embedding into the WMMA B-operand (32x16 bf16, K x N) lane
// layout, split into bf16 hi + bf16 lo residual. B k order used here (contiguous, same as A; not the ISA table order):
// lanes 0-15 hold column n = lane, K = 0..15 contiguous; lanes 16-31 hold the
// same columns, K = 16..31. Stored per (tile, chunk, lane) as 16 contiguous
// bf16 so the main loop does one 32-byte load per operand.
// ---------------------------------------------------------------------------
__global__ void vq_prep_b(const float* __restrict__ emb,
                          __bf16* __restrict__ bhi, __bf16* __restrict__ blo) {
  const int t = blockIdx.x;           // column tile (16 codes)
  const int c = blockIdx.y;           // K chunk (32 of D=64)
  const int lane = threadIdx.x;       // 0..31
  const int col = t * 16 + (lane & 15);
  const int kb  = c * 32 + ((lane >> 4) * 16);
  const float* src = emb + (size_t)col * D_DIM + kb;   // 16 contiguous floats
  const size_t dst = ((size_t)(t * 2 + c) * 32 + lane) * 16;
  #pragma unroll
  for (int j = 0; j < 16; ++j) {
    float x = src[j];
    __bf16 h = (__bf16)x;
    bhi[dst + j] = h;
    blo[dst + j] = (__bf16)(x - (float)h);
  }
}

// e2[k] = ||embedding[k]||^2
__global__ void vq_e2(const float* __restrict__ emb, float* __restrict__ e2) {
  const int k = blockIdx.x * blockDim.x + threadIdx.x;
  if (k >= K_EMB) return;
  const float4* p = (const float4*)(emb + (size_t)k * D_DIM);
  float s = 0.0f;
  #pragma unroll
  for (int j = 0; j < D_DIM / 4; ++j) {
    float4 v = p[j];
    s += v.x * v.x + v.y * v.y + v.z * v.z + v.w * v.w;
  }
  e2[k] = s;
}

// ---------------------------------------------------------------------------
// Main: each wave handles MBLK*16 = 32 rows, sweeps all 64 column tiles.
// ---------------------------------------------------------------------------
__global__ __launch_bounds__(256) void vq_main(
    const float* __restrict__ ze, const float* __restrict__ emb,
    const __bf16* __restrict__ bhi, const __bf16* __restrict__ blo,
    const float* __restrict__ e2g, float* __restrict__ cnt,
    float* __restrict__ part, float* __restrict__ out_zq,
    float* __restrict__ out_idx) {
  __shared__ float s_e2[K_EMB];
  const int tid = threadIdx.x;
  for (int i = tid; i < K_EMB; i += 256) s_e2[i] = e2g[i];
  __syncthreads();

  const int lane = tid & 31;
  const int wave = tid >> 5;
  const int m    = lane & 15;          // row (A) / col (B,C) within tile
  const int hg   = lane >> 4;          // half-wave group
  const int gw   = blockIdx.x * WAVES + wave;
  const int r0   = gw * (MBLK * 16);

  // --- load A (z rows) in WMMA A layout, bf16 hi/lo split ---------------
  // 16-bit A 16x32: lanes 0-15: K 0..15; lanes 16-31: K 16..31 (matches B)
  v16bf a_hi[MBLK][2], a_lo[MBLK][2];
  #pragma unroll
  for (int b = 0; b < MBLK; ++b) {
    const float* zr = ze + (size_t)(r0 + b * 16 + m) * D_DIM;
    #pragma unroll
    for (int c = 0; c < 2; ++c) {
      const int kb = c * 32 + hg * 16;
      float tmp[16];
      #pragma unroll
      for (int j = 0; j < 8; ++j) tmp[j]     = zr[kb + j];
      #pragma unroll
      for (int j = 0; j < 8; ++j) tmp[8 + j] = zr[kb + 8 + j];
      #pragma unroll
      for (int j = 0; j < 16; ++j) {
        float x = tmp[j];
        __bf16 h = (__bf16)x;
        a_hi[b][c][j] = h;
        a_lo[b][c][j] = (__bf16)(x - (float)h);
      }
    }
  }

  // running argmin per C-layout slot: row = v + 8*hg, col = m
  float minv[MBLK][8];
  int   mini[MBLK][8];
  #pragma unroll
  for (int b = 0; b < MBLK; ++b)
    #pragma unroll
    for (int v = 0; v < 8; ++v) { minv[b][v] = 3.402823e38f; mini[b][v] = 0; }

  for (int t = 0; t < N_CTILE; ++t) {
    v8f acc[MBLK];
    #pragma unroll
    for (int b = 0; b < MBLK; ++b) acc[b] = (v8f){0, 0, 0, 0, 0, 0, 0, 0};
    #pragma unroll
    for (int c = 0; c < 2; ++c) {
      const size_t boff = ((size_t)(t * 2 + c) * 32 + lane) * 16;
      v16bf bh = *(const v16bf*)(bhi + boff);
      v16bf bl = *(const v16bf*)(blo + boff);
      #pragma unroll
      for (int b = 0; b < MBLK; ++b) {
        acc[b] = __builtin_amdgcn_wmma_f32_16x16x32_bf16(
            false, a_hi[b][c], false, bh, (short)0, acc[b], false, false);
        acc[b] = __builtin_amdgcn_wmma_f32_16x16x32_bf16(
            false, a_hi[b][c], false, bl, (short)0, acc[b], false, false);
        acc[b] = __builtin_amdgcn_wmma_f32_16x16x32_bf16(
            false, a_lo[b][c], false, bh, (short)0, acc[b], false, false);
      }
    }
    const float e2c = s_e2[t * 16 + m];
    const int   idx = t * 16 + m;
    #pragma unroll
    for (int b = 0; b < MBLK; ++b)
      #pragma unroll
      for (int v = 0; v < 8; ++v) {
        float s = __builtin_fmaf(-2.0f, acc[b][v], e2c);  // e2 - 2*dot
        if (s < minv[b][v]) { minv[b][v] = s; mini[b][v] = idx; }
      }
  }

  // argmin reduction across the 16 lanes of each half-wave (xor 1,2,4,8
  // stays within a 16-lane group on wave32); tie-break on lower index.
  #pragma unroll
  for (int off = 1; off < 16; off <<= 1) {
    #pragma unroll
    for (int b = 0; b < MBLK; ++b)
      #pragma unroll
      for (int v = 0; v < 8; ++v) {
        float ov = __shfl_xor(minv[b][v], off, 32);
        int   oi = __shfl_xor(mini[b][v], off, 32);
        if (ov < minv[b][v] || (ov == minv[b][v] && oi < mini[b][v])) {
          minv[b][v] = ov; mini[b][v] = oi;
        }
      }
  }

  // indices (as float, per harness output dtype) + fp32 histogram (exact ints)
  if (m == 0) {
    #pragma unroll
    for (int b = 0; b < MBLK; ++b)
      #pragma unroll
      for (int v = 0; v < 8; ++v) {
        const int row = r0 + b * 16 + hg * 8 + v;
        out_idx[row] = (float)mini[b][v];
        atomicAdd(&cnt[mini[b][v]], 1.0f);
      }
  }

  // gather z_q = embedding[idx] (== z_q_st forward value) + commit partial
  float csum = 0.0f;
  #pragma unroll
  for (int b = 0; b < MBLK; ++b)
    #pragma unroll
    for (int v = 0; v < 8; ++v) {
      const int row = r0 + b * 16 + hg * 8 + v;
      const int idx = mini[b][v];  // identical across the 16-lane half
      const float4 q = *(const float4*)(emb + (size_t)idx * D_DIM + m * 4);
      const float4 z = *(const float4*)(ze  + (size_t)row * D_DIM + m * 4);
      *(float4*)(out_zq + (size_t)row * D_DIM + m * 4) = q;
      const float dx = z.x - q.x, dy = z.y - q.y;
      const float dz = z.z - q.z, dw = z.w - q.w;
      csum += dx * dx + dy * dy + dz * dz + dw * dw;
    }
  #pragma unroll
  for (int off = 1; off < 32; off <<= 1) csum += __shfl_xor(csum, off, 32);
  if (lane == 0) part[gw] = csum;   // deterministic: fixed slot per wave
}

// ---------------------------------------------------------------------------
// Final: avg_probs, perplexity, vq_loss (deterministic tree reductions).
// out_tail layout: [0]=vq_loss, [1]=perplexity, [2..2+K)=avg_probs
// ---------------------------------------------------------------------------
__global__ void vq_final(const float* __restrict__ cnt,
                         const float* __restrict__ part,
                         float* __restrict__ out_tail) {
  const int tid = threadIdx.x;
  float s = 0.0f, c = 0.0f;
  for (int k = tid; k < K_EMB; k += 256) {
    const float p = cnt[k] * (1.0f / (float)N_PTS);
    out_tail[2 + k] = p;
    s += p * __logf(p + 1e-10f);
  }
  for (int i = tid; i < N_WAVES_TOTAL; i += 256) c += part[i];
  #pragma unroll
  for (int off = 1; off < 32; off <<= 1) {
    s += __shfl_xor(s, off, 32);
    c += __shfl_xor(c, off, 32);
  }
  __shared__ float sm[8], cm[8];
  if ((tid & 31) == 0) { sm[tid >> 5] = s; cm[tid >> 5] = c; }
  __syncthreads();
  if (tid == 0) {
    float st = 0.0f, ct = 0.0f;
    #pragma unroll
    for (int w = 0; w < 8; ++w) { st += sm[w]; ct += cm[w]; }
    out_tail[0] = BETA_C * ct / (float)(N_PTS * D_DIM);  // vq_loss
    out_tail[1] = __expf(-st);                           // perplexity
  }
}

// ---------------------------------------------------------------------------
extern "C" void kernel_launch(void* const* d_in, const int* in_sizes, int n_in,
                              void* d_out, int out_size, void* d_ws,
                              size_t ws_size, hipStream_t stream) {
  const float* ze  = (const float*)d_in[0];   // (131072, 64) f32
  const float* emb = (const float*)d_in[1];   // (1024, 64)  f32
  float* out = (float*)d_out;
  char*  ws  = (char*)d_ws;

  __bf16* bhi = (__bf16*)(ws + WS_BHI);
  __bf16* blo = (__bf16*)(ws + WS_BLO);
  float*  e2  = (float*)(ws + WS_E2);
  float*  cnt = (float*)(ws + WS_CNT);
  float*  prt = (float*)(ws + WS_PART);

  float* out_zq   = out;                                  // N*D
  float* out_idx  = out + (size_t)N_PTS * D_DIM;          // N
  float* out_tail = out_idx + N_PTS;                      // 1+1+K

  hipMemsetAsync(cnt, 0, K_EMB * sizeof(float), stream);
  vq_prep_b<<<dim3(N_CTILE, 2), 32, 0, stream>>>(emb, bhi, blo);
  vq_e2<<<K_EMB / 256, 256, 0, stream>>>(emb, e2);
  vq_main<<<N_BLOCKS, 256, 0, stream>>>(ze, emb, bhi, blo, e2, cnt, prt,
                                        out_zq, out_idx);
  vq_final<<<1, 256, 0, stream>>>(cnt, prt, out_tail);
}
